// MegaCRN_5471788335942
// MI455X (gfx1250) — compile-verified, matrix operand order checked by reading, not run on hardware
//
#include <hip/hip_runtime.h>
#include <hip/hip_bf16.h>
#include <math.h>

// ---------------- model dims ----------------
#define B_    16
#define T_    12
#define N_    512
#define HZ_   12
#define RNN_  64
#define MEMN_ 20
#define MEMD_ 64
#define S_    6          // 2*K supports
#define DDEC_ 128
#define CPE_  80         // padded enc channels (65 -> 80)
#define CPD_  144        // padded dec channels (130 -> 144)
#define NN_   (N_*N_)
#define BN_   (B_*N_)
#define EFLEN 524288     // e_flat row length = B*T*N*RNN / T
#define NBLK  256

#define ACT_NONE 0
#define ACT_SIG  1
#define ACT_TANH 2
#define ACT_RELU 3

typedef __attribute__((ext_vector_type(16))) _Float16 v16h;
typedef __attribute__((ext_vector_type(4)))  _Float16 v4h;
typedef __attribute__((ext_vector_type(8)))  float    v8f;

union FragU { v16h v; unsigned u[8]; };

// ---------------- WMMA GEMM ----------------
// C = act(A @ B(^T) + bias).
// REQUIREMENTS (all call sites obey): M % 128 == 0, K % 32 == 0,
// lda/ldb % 4 == 0, base pointers 16B aligned. Only N may be ragged.
// Software pipeline: register prefetch of tile k+1 overlaps the 4 WMMAs of
// tile k; double-buffered LDS -> one barrier per K-step.
#define TM 128
#define TN 64
#define TK 32
#define LDS_K (TK + 8)

__global__ __launch_bounds__(256)
void k_wmma_gemm(const float* __restrict__ A, const float* __restrict__ Bm,
                 float* __restrict__ C, const float* __restrict__ bias,
                 int M, int Ncols, int K, int lda, int ldb, int ldc,
                 int transB, int act, int S,
                 long sAk, long sBb, long sCb, long sCk)
{
    int bz = blockIdx.z;
    int kb = bz % S, bb = bz / S;
    A  += (long)kb * sAk;
    Bm += (long)bb * sBb;
    C  += (long)bb * sCb + (long)kb * sCk;

    __shared__ _Float16 As [2][TM][LDS_K];   // m-major, k-contiguous
    __shared__ _Float16 BsT[2][TN][LDS_K];   // n-major, k-contiguous

    int tid   = threadIdx.x;
    int tileM = blockIdx.y * TM;
    int tileN = blockIdx.x * TN;
    int w     = tid >> 5;
    int wm    = w >> 1, wn = w & 1;          // 4 x 2 wave grid, 32x32 per wave
    int lane  = tid & 31;
    int hh    = lane >> 4;
    int lrow  = lane & 15;

    bool fullN = (tileN + TN) <= Ncols;

    v8f acc00 = {}, acc01 = {}, acc10 = {}, acc11 = {};

    float4 ra[4], rb[2];

    // ---- issue all global loads for K-slice k0 into registers ----
    auto glb = [&](int k0) {
        #pragma unroll
        for (int i = 0; i < 4; ++i) {
            int l = i * 256 + tid;
            int r = l >> 3, c4 = (l & 7) << 2;
            ra[i] = *(const float4*)(A + (long)(tileM + r) * lda + (k0 + c4));
        }
        if (transB) {
            #pragma unroll
            for (int i = 0; i < 2; ++i) {
                int l = i * 256 + tid;
                int n = l >> 3, k4 = (l & 7) << 2;
                int gn = tileN + n;
                if (fullN || gn < Ncols) {
                    rb[i] = *(const float4*)(Bm + (long)gn * ldb + (k0 + k4));
                } else {
                    rb[i].x = rb[i].y = rb[i].z = rb[i].w = 0.f;
                }
            }
        } else {
            #pragma unroll
            for (int i = 0; i < 2; ++i) {
                int l = i * 256 + tid;
                int kr = l >> 4, n4 = (l & 15) << 2;
                const float* g = Bm + (long)(k0 + kr) * ldb + (tileN + n4);
                if (fullN) {
                    rb[i] = *(const float4*)g;
                } else {
                    float4 v;
                    v.x = (tileN + n4 + 0 < Ncols) ? g[0] : 0.f;
                    v.y = (tileN + n4 + 1 < Ncols) ? g[1] : 0.f;
                    v.z = (tileN + n4 + 2 < Ncols) ? g[2] : 0.f;
                    v.w = (tileN + n4 + 3 < Ncols) ? g[3] : 0.f;
                    rb[i] = v;
                }
            }
        }
    };

    // ---- convert and stash registers into LDS buffer ----
    auto stash = [&](int buf) {
        #pragma unroll
        for (int i = 0; i < 4; ++i) {
            int l = i * 256 + tid;
            int r = l >> 3, c4 = (l & 7) << 2;
            v4h h; h[0] = (_Float16)ra[i].x; h[1] = (_Float16)ra[i].y;
                   h[2] = (_Float16)ra[i].z; h[3] = (_Float16)ra[i].w;
            *(v4h*)&As[buf][r][c4] = h;
        }
        if (transB) {
            #pragma unroll
            for (int i = 0; i < 2; ++i) {
                int l = i * 256 + tid;
                int n = l >> 3, k4 = (l & 7) << 2;
                v4h h; h[0] = (_Float16)rb[i].x; h[1] = (_Float16)rb[i].y;
                       h[2] = (_Float16)rb[i].z; h[3] = (_Float16)rb[i].w;
                *(v4h*)&BsT[buf][n][k4] = h;
            }
        } else {
            #pragma unroll
            for (int i = 0; i < 2; ++i) {
                int l = i * 256 + tid;
                int kr = l >> 4, n4 = (l & 15) << 2;
                BsT[buf][n4 + 0][kr] = (_Float16)rb[i].x;
                BsT[buf][n4 + 1][kr] = (_Float16)rb[i].y;
                BsT[buf][n4 + 2][kr] = (_Float16)rb[i].z;
                BsT[buf][n4 + 3][kr] = (_Float16)rb[i].w;
            }
        }
    };

    glb(0);
    stash(0);
    __syncthreads();

    int cur = 0;
    for (int k0 = 0; k0 < K; k0 += TK) {
        bool more = (k0 + TK) < K;
        if (more) glb(k0 + TK);          // prefetch: loads fly behind WMMAs

        FragU a0, a1, b0, b1;
        {
            const unsigned* r0 = (const unsigned*)&As [cur][wm * 32 + lrow][0];
            const unsigned* r1 = (const unsigned*)&As [cur][wm * 32 + 16 + lrow][0];
            const unsigned* c0 = (const unsigned*)&BsT[cur][wn * 32 + lrow][0];
            const unsigned* c1 = (const unsigned*)&BsT[cur][wn * 32 + 16 + lrow][0];
            #pragma unroll
            for (int i = 0; i < 4; ++i) {
                a0.u[i] = r0[hh * 4 + i];  a0.u[4 + i] = r0[8 + hh * 4 + i];
                a1.u[i] = r1[hh * 4 + i];  a1.u[4 + i] = r1[8 + hh * 4 + i];
                b0.u[i] = c0[hh * 4 + i];  b0.u[4 + i] = c0[8 + hh * 4 + i];
                b1.u[i] = c1[hh * 4 + i];  b1.u[4 + i] = c1[8 + hh * 4 + i];
            }
        }
        acc00 = __builtin_amdgcn_wmma_f32_16x16x32_f16(false, a0.v, false, b0.v, (short)0, acc00, false, false);
        acc01 = __builtin_amdgcn_wmma_f32_16x16x32_f16(false, a0.v, false, b1.v, (short)0, acc01, false, false);
        acc10 = __builtin_amdgcn_wmma_f32_16x16x32_f16(false, a1.v, false, b0.v, (short)0, acc10, false, false);
        acc11 = __builtin_amdgcn_wmma_f32_16x16x32_f16(false, a1.v, false, b1.v, (short)0, acc11, false, false);

        if (more) {
            stash(cur ^ 1);
            __syncthreads();             // single barrier per K-step
        }
        cur ^= 1;
    }

    // ---- epilogue ----
    v8f* accs[2][2] = { { &acc00, &acc01 }, { &acc10, &acc11 } };
    #pragma unroll
    for (int ni = 0; ni < 2; ++ni) {
        int gn = tileN + wn * 32 + ni * 16 + lrow;
        if (gn >= Ncols) continue;
        float bv = bias ? bias[gn] : 0.f;
        #pragma unroll
        for (int mi = 0; mi < 2; ++mi) {
            v8f& a = *accs[mi][ni];
            #pragma unroll
            for (int v = 0; v < 8; ++v) {
                int gm = tileM + wm * 32 + mi * 16 + (hh ? v + 8 : v);
                float x = a[v] + bv;
                if      (act == ACT_SIG)  x = 1.f / (1.f + expf(-x));
                else if (act == ACT_TANH) x = tanhf(x);
                else if (act == ACT_RELU) x = fmaxf(x, 0.f);
                C[(long)gm * ldc + gn] = x;
            }
        }
    }
}

// ---------------- scalar GEMM (tiny K or N) ----------------
__global__ void k_small_gemm(const float* __restrict__ A, const float* __restrict__ Bm,
                             float* __restrict__ C, const float* __restrict__ bias,
                             int M, int Ncols, int K, int lda, int ldb, int ldc,
                             int transB, int act)
{
    long i = (long)blockIdx.x * 256 + threadIdx.x;
    if (i >= (long)M * Ncols) return;
    int m = (int)(i / Ncols), n = (int)(i % Ncols);
    float s = bias ? bias[n] : 0.f;
    for (int k = 0; k < K; ++k)
        s += A[(long)m * lda + k] * (transB ? Bm[(long)n * ldb + k]
                                            : Bm[(long)k * ldb + n]);
    if      (act == ACT_SIG)  s = 1.f / (1.f + expf(-s));
    else if (act == ACT_TANH) s = tanhf(s);
    else if (act == ACT_RELU) s = fmaxf(s, 0.f);
    C[(long)m * ldc + n] = s;
}

// ---------------- row softmax (in place) ----------------
__global__ void k_softmax_rows(float* X, int cols, int ld)
{
    __shared__ float red[256];
    float* p = X + (long)blockIdx.x * ld;
    int tid = threadIdx.x;
    float m = -1e30f;
    for (int c = tid; c < cols; c += 256) m = fmaxf(m, p[c]);
    red[tid] = m; __syncthreads();
    for (int s = 128; s > 0; s >>= 1) { if (tid < s) red[tid] = fmaxf(red[tid], red[tid+s]); __syncthreads(); }
    m = red[0]; __syncthreads();
    float sum = 0.f;
    for (int c = tid; c < cols; c += 256) { float e = expf(p[c] - m); p[c] = e; sum += e; }
    red[tid] = sum; __syncthreads();
    for (int s = 128; s > 0; s >>= 1) { if (tid < s) red[tid] += red[tid+s]; __syncthreads(); }
    float inv = 1.f / red[0];
    for (int c = tid; c < cols; c += 256) p[c] *= inv;
}

// ---------------- misc elementwise ----------------
__global__ void k_set_identity(float* P, int n)
{
    long i = (long)blockIdx.x * 256 + threadIdx.x;
    if (i >= (long)n * n) return;
    P[i] = (i / n == i % n) ? 1.f : 0.f;
}

__global__ void k_two_x_minus_I(float* P, int n)
{
    long i = (long)blockIdx.x * 256 + threadIdx.x;
    if (i >= (long)n * n) return;
    P[i] = 2.f * P[i] - ((i / n == i % n) ? 1.f : 0.f);
}

__global__ void k_pad_weight(const float* src, float* dst, int S, int Cin, int CP, int O)
{
    long i = (long)blockIdx.x * 256 + threadIdx.x;
    if (i >= (long)S * CP * O) return;
    int o = (int)(i % O); long rc = i / O; int c = (int)(rc % CP); int k = (int)(rc / CP);
    dst[i] = (c < Cin) ? src[((long)k * Cin + c) * O + o] : 0.f;
}

__global__ void k_build_ins_enc(const float* x, int t, const float* state,
                                const float* zr, float* insP)
{
    long i = (long)blockIdx.x * 256 + threadIdx.x;
    if (i >= (long)BN_ * CPE_) return;
    int c = (int)(i % CPE_); long bn = i / CPE_;
    int b = (int)(bn / N_), n = (int)(bn % N_);
    float v = 0.f;
    if (c == 0) v = x[((long)b * T_ + t) * N_ + n];
    else if (c <= RNN_) {
        v = state[bn * RNN_ + (c - 1)];
        if (zr) v *= zr[bn * (2 * RNN_) + (c - 1)];
    }
    insP[i] = v;
}

__global__ void k_build_ins_dec(const float* go, const float* ycov, int t,
                                const float* state, const float* zr, float* insP)
{
    long i = (long)blockIdx.x * 256 + threadIdx.x;
    if (i >= (long)BN_ * CPD_) return;
    int c = (int)(i % CPD_); long bn = i / CPD_;
    int b = (int)(bn / N_), n = (int)(bn % N_);
    float v = 0.f;
    if (c == 0) v = go[bn];
    else if (c == 1) v = ycov[((long)b * HZ_ + t) * N_ + n];
    else if (c < 2 + DDEC_) {
        v = state[bn * DDEC_ + (c - 2)];
        if (zr) v *= zr[bn * (2 * DDEC_) + (c - 2)];
    }
    insP[i] = v;
}

__global__ void k_cell_combine(float* state, const float* zr, const float* hc, int H)
{
    long i = (long)blockIdx.x * 256 + threadIdx.x;
    if (i >= (long)BN_ * H) return;
    long bn = i / H; int h = (int)(i % H);
    float r = zr[bn * (2 * H) + H + h];
    state[i] = r * state[i] + (1.f - r) * hc[i];
}

__global__ void k_store_hen(const float* state, float* h_en, int t)
{
    long i = (long)blockIdx.x * 256 + threadIdx.x;
    if (i >= (long)BN_ * RNN_) return;
    long bn = i / RNN_; int r = (int)(i % RNN_);
    int b = (int)(bn / N_), n = (int)(bn % N_);
    h_en[(((long)b * T_ + t) * N_ + n) * RNN_ + r] = state[i];
}

__global__ void k_concat_ht(const float* ht, const float* hatt, float* sd)
{
    long i = (long)blockIdx.x * 256 + threadIdx.x;
    if (i >= (long)BN_ * DDEC_) return;
    long bn = i / DDEC_; int c = (int)(i % DDEC_);
    sd[i] = (c < RNN_) ? ht[bn * RNN_ + c] : hatt[bn * MEMD_ + (c - RNN_)];
}

__global__ void k_topk_gather(const float* att, const float* Mem, float* pos, float* neg)
{
    long bn = (long)blockIdx.x * 256 + threadIdx.x;
    if (bn >= BN_) return;
    float sc[MEMN_]; int idx[MEMN_];
    for (int j = 0; j < MEMN_; ++j) { sc[j] = att[bn * MEMN_ + j]; idx[j] = j; }
    for (int a = 0; a < MEMN_; ++a) {           // selection sort desc (swap-based, not stable)
        int best = a;
        for (int b2 = a + 1; b2 < MEMN_; ++b2)
            if (sc[b2] > sc[best]) best = b2;
        float ts = sc[a]; sc[a] = sc[best]; sc[best] = ts;
        int   ti = idx[a]; idx[a] = idx[best]; idx[best] = ti;
    }
    for (int d = 0; d < MEMD_; ++d) pos[bn * MEMD_ + d] = Mem[idx[0] * MEMD_ + d];
    for (int j = 1; j < MEMN_; ++j)
        for (int d = 0; d < MEMD_; ++d)
            neg[(bn * (MEMN_ - 1) + (j - 1)) * MEMD_ + d] = Mem[idx[j] * MEMD_ + d];
}

__global__ void k_dot_partial(const float* __restrict__ E, const float* __restrict__ D,
                              float* part)
{
    __shared__ float red[256];
    int row = blockIdx.y, blk = blockIdx.x, tid = threadIdx.x;
    long base = (long)row * EFLEN + (long)blk * 2048;
    float s = 0.f;
    #pragma unroll
    for (int j = 0; j < 8; ++j) {
        long o = (long)blk * 2048 + j * 256 + tid;
        s += E[base + j * 256 + tid] * D[o];
    }
    red[tid] = s; __syncthreads();
    for (int st = 128; st > 0; st >>= 1) { if (tid < st) red[tid] += red[tid + st]; __syncthreads(); }
    if (tid == 0) part[row * NBLK + blk] = red[0];
}

__global__ void k_att_finalize(const float* part, float* attv)
{
    __shared__ float red[256];
    __shared__ float s[T_];
    int tid = threadIdx.x;
    for (int r = 0; r < T_; ++r) {
        red[tid] = part[r * NBLK + tid];
        __syncthreads();
        for (int st = 128; st > 0; st >>= 1) { if (tid < st) red[tid] += red[tid + st]; __syncthreads(); }
        if (tid == 0) s[r] = red[0];
        __syncthreads();
    }
    if (tid == 0) {
        float m = -1e30f;
        for (int r = 0; r < T_; ++r) m = fmaxf(m, s[r]);
        float sum = 0.f;
        for (int r = 0; r < T_; ++r) { s[r] = expf(s[r] - m); sum += s[r]; }
        for (int r = 0; r < T_; ++r) attv[r] = s[r] / sum;
    }
}

__global__ void k_context(const float* __restrict__ h_en, const float* attv, float* ctx)
{
    long i = (long)blockIdx.x * 256 + threadIdx.x;
    if (i >= (long)BN_ * RNN_) return;
    int r = (int)(i % RNN_); long bn = i / RNN_;
    int n = (int)(bn % N_), b = (int)(bn / N_);
    float s = 0.f;
    for (int t = 0; t < T_; ++t)
        s += h_en[(((long)b * T_ + t) * N_ + n) * RNN_ + r] * attv[t];
    ctx[i] = s;
}

__global__ void k_concat_cat(const float* hde, const float* ctx, float* cat)
{
    long i = (long)blockIdx.x * 256 + threadIdx.x;
    if (i >= (long)BN_ * (DDEC_ + RNN_)) return;
    long bn = i / (DDEC_ + RNN_); int c = (int)(i % (DDEC_ + RNN_));
    cat[i] = (c < DDEC_) ? hde[bn * DDEC_ + c] : ctx[bn * RNN_ + (c - DDEC_)];
}

__global__ void k_proj(const float* cat, const float* W, const float* bconst,
                       float* go, float* out, int t)
{
    long bn = (long)blockIdx.x * 256 + threadIdx.x;
    if (bn >= BN_) return;
    float s = bconst[0];
    for (int k = 0; k < DDEC_ + RNN_; ++k) s += cat[bn * (DDEC_ + RNN_) + k] * W[k];
    go[bn] = s;
    int b = (int)(bn / N_), n = (int)(bn % N_);
    out[((long)b * HZ_ + t) * N_ + n] = s;
}

__global__ void k_zero(float* p, long n)
{
    long i = (long)blockIdx.x * 256 + threadIdx.x;
    if (i < n) p[i] = 0.f;
}

// ---------------- host orchestration ----------------
static inline dim3 g1(long n) { return dim3((unsigned)((n + 255) / 256)); }

extern "C" void kernel_launch(void* const* d_in, const int* in_sizes, int n_in,
                              void* d_out, int out_size, void* d_ws, size_t ws_size,
                              hipStream_t stream)
{
    const float* x      = (const float*)d_in[0];
    const float* ycov   = (const float*)d_in[1];
    const float* Memory = (const float*)d_in[2];
    const float* Wq     = (const float*)d_in[3];
    const float* We1    = (const float*)d_in[4];
    const float* We2    = (const float*)d_in[5];
    const float* tWe1   = (const float*)d_in[6];
    const float* tWe2   = (const float*)d_in[7];
    const float* fc_d_W = (const float*)d_in[8];
    const float* fc_e_W = (const float*)d_in[9];
    const float* egW    = (const float*)d_in[10];
    const float* egb    = (const float*)d_in[11];
    const float* euW    = (const float*)d_in[12];
    const float* eub    = (const float*)d_in[13];
    const float* dgW    = (const float*)d_in[14];
    const float* dgb    = (const float*)d_in[15];
    const float* duW    = (const float*)d_in[16];
    const float* dub    = (const float*)d_in[17];
    const float* projW  = (const float*)d_in[18];
    const float* projb  = (const float*)d_in[19];

    float* out      = (float*)d_out;
    float* o_output = out;                                   // (B,HZ,N,1)
    float* o_hatt   = o_output + (long)B_*HZ_*N_;            // (B,N,64)
    float* o_query  = o_hatt   + (long)BN_*MEMD_;            // (B,N,64)
    float* o_pos    = o_query  + (long)BN_*MEMD_;            // (B,N,64)
    float* o_neg    = o_pos    + (long)BN_*MEMD_;            // (B,N,19,64)
    float* o_attsc  = o_neg    + (long)BN_*(MEMN_-1)*MEMD_;  // (B,N,20)

    float* w = (float*)d_ws;
    size_t off = 0;
    auto alloc = [&](size_t n) { float* p = w + off; off += n; return p; };
    float* sup     = alloc((size_t)S_ * NN_);
    float* tsup    = alloc((size_t)S_ * NN_);
    float* ne1     = alloc((size_t)N_ * MEMD_);
    float* ne2     = alloc((size_t)N_ * MEMD_);
    float* ne3     = alloc((size_t)N_ * RNN_);
    float* ne4     = alloc((size_t)N_ * RNN_);
    float* WgPe    = alloc((size_t)S_ * CPE_ * 2 * RNN_);
    float* WuPe    = alloc((size_t)S_ * CPE_ * RNN_);
    float* WgPd    = alloc((size_t)S_ * CPD_ * 2 * DDEC_);
    float* WuPd    = alloc((size_t)S_ * CPD_ * DDEC_);
    float* insP    = alloc((size_t)BN_ * CPD_);
    float* XG      = alloc((size_t)BN_ * S_ * CPD_);
    float* ZR      = alloc((size_t)BN_ * 2 * DDEC_);
    float* HC      = alloc((size_t)BN_ * DDEC_);
    float* state_e = alloc((size_t)BN_ * RNN_);
    float* state_d = alloc((size_t)BN_ * DDEC_);
    float* h_en    = alloc((size_t)B_ * T_ * N_ * RNN_);
    float* E       = alloc((size_t)B_ * T_ * N_ * RNN_);
    float* Dflat   = alloc((size_t)BN_ * RNN_);
    float* part    = alloc((size_t)T_ * NBLK);
    float* attv    = alloc(16);
    float* ctx     = alloc((size_t)BN_ * RNN_);
    float* cat     = alloc((size_t)BN_ * (DDEC_ + RNN_));
    float* go      = alloc((size_t)BN_);

    auto gemm = [&](const float* A, const float* Bm, float* C, const float* bias,
                    int M, int Nc, int K, int lda, int ldb, int ldc,
                    int transB, int act, int nbatch, int Sdiv,
                    long sAk, long sBb, long sCb, long sCk) {
        dim3 grid((Nc + TN - 1) / TN, (M + TM - 1) / TM, nbatch);
        k_wmma_gemm<<<grid, 256, 0, stream>>>(A, Bm, C, bias, M, Nc, K, lda, ldb, ldc,
                                              transB, act, Sdiv, sAk, sBb, sCb, sCk);
    };

    auto build_graph = [&](const float* na, const float* nb, int kdim, float* sb) {
        k_set_identity<<<g1(NN_), 256, 0, stream>>>(sb, N_);
        k_set_identity<<<g1(NN_), 256, 0, stream>>>(sb + 3L * NN_, N_);
        gemm(na, nb, sb + 1L * NN_, nullptr, N_, N_, kdim, kdim, kdim, N_, 1, ACT_RELU, 1, 1, 0,0,0,0);
        k_softmax_rows<<<N_, 256, 0, stream>>>(sb + 1L * NN_, N_, N_);
        gemm(nb, na, sb + 4L * NN_, nullptr, N_, N_, kdim, kdim, kdim, N_, 1, ACT_RELU, 1, 1, 0,0,0,0);
        k_softmax_rows<<<N_, 256, 0, stream>>>(sb + 4L * NN_, N_, N_);
        gemm(sb + 1L*NN_, sb + 1L*NN_, sb + 2L*NN_, nullptr, N_, N_, N_, N_, N_, N_, 0, ACT_NONE, 1, 1, 0,0,0,0);
        k_two_x_minus_I<<<g1(NN_), 256, 0, stream>>>(sb + 2L * NN_, N_);
        gemm(sb + 4L*NN_, sb + 4L*NN_, sb + 5L*NN_, nullptr, N_, N_, N_, N_, N_, N_, 0, ACT_NONE, 1, 1, 0,0,0,0);
        k_two_x_minus_I<<<g1(NN_), 256, 0, stream>>>(sb + 5L * NN_, N_);
    };

    // ---- padded weights (once) ----
    k_pad_weight<<<g1((long)S_*CPE_*2*RNN_), 256, 0, stream>>>(egW, WgPe, S_, 65, CPE_, 2*RNN_);
    k_pad_weight<<<g1((long)S_*CPE_*RNN_),   256, 0, stream>>>(euW, WuPe, S_, 65, CPE_, RNN_);
    k_pad_weight<<<g1((long)S_*CPD_*2*DDEC_),256, 0, stream>>>(dgW, WgPd, S_, 130, CPD_, 2*DDEC_);
    k_pad_weight<<<g1((long)S_*CPD_*DDEC_),  256, 0, stream>>>(duW, WuPd, S_, 130, CPD_, DDEC_);

    // ---- memory graph (sup) ----
    k_small_gemm<<<g1((long)N_*MEMD_), 256, 0, stream>>>(We1, Memory, ne1, nullptr, N_, MEMD_, MEMN_, MEMN_, MEMD_, MEMD_, 0, ACT_NONE);
    k_small_gemm<<<g1((long)N_*MEMD_), 256, 0, stream>>>(We2, Memory, ne2, nullptr, N_, MEMD_, MEMN_, MEMN_, MEMD_, MEMD_, 0, ACT_NONE);
    build_graph(ne1, ne2, MEMD_, sup);

    // ---- encoder ----
    k_zero<<<g1((long)BN_*RNN_), 256, 0, stream>>>(state_e, (long)BN_*RNN_);
    for (int t = 0; t < T_; ++t) {
        k_build_ins_enc<<<g1((long)BN_*CPE_), 256, 0, stream>>>(x, t, state_e, nullptr, insP);
        gemm(sup, insP, XG, nullptr, N_, CPE_, N_, N_, CPE_, S_*CPE_, 0, ACT_NONE,
             S_*B_, S_, NN_, (long)N_*CPE_, (long)N_*S_*CPE_, CPE_);
        gemm(XG, WgPe, ZR, egb, BN_, 2*RNN_, S_*CPE_, S_*CPE_, 2*RNN_, 2*RNN_, 0, ACT_SIG, 1, 1, 0,0,0,0);
        k_build_ins_enc<<<g1((long)BN_*CPE_), 256, 0, stream>>>(x, t, state_e, ZR, insP);
        gemm(sup, insP, XG, nullptr, N_, CPE_, N_, N_, CPE_, S_*CPE_, 0, ACT_NONE,
             S_*B_, S_, NN_, (long)N_*CPE_, (long)N_*S_*CPE_, CPE_);
        gemm(XG, WuPe, HC, eub, BN_, RNN_, S_*CPE_, S_*CPE_, RNN_, RNN_, 0, ACT_TANH, 1, 1, 0,0,0,0);
        k_cell_combine<<<g1((long)BN_*RNN_), 256, 0, stream>>>(state_e, ZR, HC, RNN_);
        k_store_hen<<<g1((long)BN_*RNN_), 256, 0, stream>>>(state_e, h_en, t);
    }

    // ---- memory attention ----
    gemm(state_e, Wq, o_query, nullptr, BN_, MEMD_, RNN_, RNN_, MEMD_, MEMD_, 0, ACT_NONE, 1, 1, 0,0,0,0);
    gemm(o_query, Memory, o_attsc, nullptr, BN_, MEMN_, MEMD_, MEMD_, MEMD_, MEMN_, 1, ACT_NONE, 1, 1, 0,0,0,0);
    k_softmax_rows<<<BN_, 256, 0, stream>>>(o_attsc, MEMN_, MEMN_);
    k_small_gemm<<<g1((long)BN_*MEMD_), 256, 0, stream>>>(o_attsc, Memory, o_hatt, nullptr, BN_, MEMD_, MEMN_, MEMN_, MEMD_, MEMD_, 0, ACT_NONE);
    k_topk_gather<<<g1(BN_), 256, 0, stream>>>(o_attsc, Memory, o_pos, o_neg);

    // ---- decoder graph (tsup) from h_att[-1] ----
    const float* hatt_last = o_hatt + 15L * N_ * MEMD_;
    gemm(hatt_last, tWe1, ne3, nullptr, N_, RNN_, MEMD_, MEMD_, RNN_, RNN_, 0, ACT_NONE, 1, 1, 0,0,0,0);
    gemm(hatt_last, tWe2, ne4, nullptr, N_, RNN_, MEMD_, MEMD_, RNN_, RNN_, 0, ACT_NONE, 1, 1, 0,0,0,0);
    build_graph(ne3, ne4, RNN_, tsup);

    // ---- e_flat = (h_en @ fc_e_W) ----
    gemm(h_en, fc_e_W, E, nullptr, B_*T_*N_, RNN_, RNN_, RNN_, RNN_, RNN_, 0, ACT_NONE, 1, 1, 0,0,0,0);

    // ---- decoder ----
    k_concat_ht<<<g1((long)BN_*DDEC_), 256, 0, stream>>>(state_e, o_hatt, state_d);
    k_zero<<<g1(BN_), 256, 0, stream>>>(go, BN_);
    for (int t = 0; t < HZ_; ++t) {
        k_build_ins_dec<<<g1((long)BN_*CPD_), 256, 0, stream>>>(go, ycov, t, state_d, nullptr, insP);
        gemm(tsup, insP, XG, nullptr, N_, CPD_, N_, N_, CPD_, S_*CPD_, 0, ACT_NONE,
             S_*B_, S_, NN_, (long)N_*CPD_, (long)N_*S_*CPD_, CPD_);
        gemm(XG, WgPd, ZR, dgb, BN_, 2*DDEC_, S_*CPD_, S_*CPD_, 2*DDEC_, 2*DDEC_, 0, ACT_SIG, 1, 1, 0,0,0,0);
        k_build_ins_dec<<<g1((long)BN_*CPD_), 256, 0, stream>>>(go, ycov, t, state_d, ZR, insP);
        gemm(tsup, insP, XG, nullptr, N_, CPD_, N_, N_, CPD_, S_*CPD_, 0, ACT_NONE,
             S_*B_, S_, NN_, (long)N_*CPD_, (long)N_*S_*CPD_, CPD_);
        gemm(XG, WuPd, HC, dub, BN_, DDEC_, S_*CPD_, S_*CPD_, DDEC_, DDEC_, 0, ACT_TANH, 1, 1, 0,0,0,0);
        k_cell_combine<<<g1((long)BN_*DDEC_), 256, 0, stream>>>(state_d, ZR, HC, DDEC_);

        gemm(state_d, fc_d_W, Dflat, nullptr, BN_, RNN_, DDEC_, DDEC_, RNN_, RNN_, 0, ACT_NONE, 1, 1, 0,0,0,0);
        k_dot_partial<<<dim3(NBLK, T_), 256, 0, stream>>>(E, Dflat, part);
        k_att_finalize<<<1, 256, 0, stream>>>(part, attv);
        k_context<<<g1((long)BN_*RNN_), 256, 0, stream>>>(h_en, attv, ctx);
        k_concat_cat<<<g1((long)BN_*(DDEC_+RNN_)), 256, 0, stream>>>(state_d, ctx, cat);
        k_proj<<<g1(BN_), 256, 0, stream>>>(cat, projW, projb, go, o_output, t);
    }
}
